// DCN_cross_interaction_88682484727829
// MI455X (gfx1250) — compile-verified, matrix operand order checked by reading, not run on hardware
//
#include <hip/hip_runtime.h>

// ---------------------------------------------------------------------------
// B=16, NQ=512, NC=4096, D=256, fp32 in.
// Outputs (flat): A_Q [16,512,4096], A_D [16,4096,512], A_Qv [16,4096], A_Dv [16,512].
//   1. L2-normalize Q/D rows -> f16 (elements <=1 -> f16 + fp32 WMMA acc accurate).
//   2. WMMA f16 GEMM writes L = Qn*Dn^T into the A_Q output region.
//      A/B k-slabs staged to LDS by the Tensor Data Mover (tensor_load_to_lds,
//      TENSORcnt-synchronized); 6-arg builtin form for this toolchain.
//   3. Column softmax reads L -> A_D (LDS-transposed, coalesced writes).
//   4. Row softmax converts the A_Q region in place.
//   5. Mean kernels produce the two vectors.
// HBM-bound on ~268 MB of output; GEMM (17.2 GFLOP) rides v_wmma_f32_16x16x32_f16.
// ---------------------------------------------------------------------------

typedef __attribute__((ext_vector_type(16))) _Float16 v16h;
typedef __attribute__((ext_vector_type(8)))  _Float16 v8h;
typedef __attribute__((ext_vector_type(8)))  float    v8f;
typedef __attribute__((ext_vector_type(4)))  float    v4f;
typedef __attribute__((ext_vector_type(4)))  int      v4i;
typedef __attribute__((ext_vector_type(8)))  int      v8i;
typedef __attribute__((ext_vector_type(4)))  unsigned int v4u;

#define BATCH 16
#define NQ    512
#define NC    4096
#define DIM   256

#if defined(__gfx1250__) && __has_builtin(__builtin_amdgcn_tensor_load_to_lds)
#define USE_TDM 1
#else
#define USE_TDM 0
#endif

// ---------------------------------------------------------------------------
// 1) Row L2-normalize fp32 [rows,256] -> f16 [rows,256]. One wave per row.
// ---------------------------------------------------------------------------
__global__ __launch_bounds__(256) void normalize_kernel(const float* __restrict__ X,
                                                        _Float16* __restrict__ Y,
                                                        int nrows) {
    const int lane = threadIdx.x & 31;
    const int wave = threadIdx.x >> 5;
    const int row  = blockIdx.x * 8 + wave;
    if (row >= nrows) return;

    const float* xp = X + (size_t)row * DIM + lane * 8;
    v4f a = *(const v4f*)(xp);
    v4f b = *(const v4f*)(xp + 4);

    float ss = a.x * a.x + a.y * a.y + a.z * a.z + a.w * a.w +
               b.x * b.x + b.y * b.y + b.z * b.z + b.w * b.w;
    #pragma unroll
    for (int off = 16; off >= 1; off >>= 1)
        ss += __shfl_xor(ss, off, 32);

    const float s = 1.0f / fmaxf(sqrtf(ss), 1e-8f);

    v8h o;
    o[0] = (_Float16)(a.x * s); o[1] = (_Float16)(a.y * s);
    o[2] = (_Float16)(a.z * s); o[3] = (_Float16)(a.w * s);
    o[4] = (_Float16)(b.x * s); o[5] = (_Float16)(b.y * s);
    o[6] = (_Float16)(b.z * s); o[7] = (_Float16)(b.w * s);
    *(v8h*)(Y + (size_t)row * DIM + lane * 8) = o;
}

// ---------------------------------------------------------------------------
// 2) WMMA GEMM: L[b] = Qh[b] (512x256) * Dh[b]^T (256x4096), fp32 out.
//    WG tile 128x128, 8 waves, wave tile 64x32 (4x2 WMMA 16x16 tiles),
//    K staged in LDS as 128x32-element (64 B/row) slabs, double buffered.
// ---------------------------------------------------------------------------
#define TM 128
#define TN 128
#define TK 32

#if USE_TDM
// One TDM descriptor moves a whole 128-row x 32-element (f16) slab:
//   tile_dim0 = 32 elems (64 B, mult of 4 B), tile_dim1 = 128 rows,
//   tensor_dim0_stride = 256 elems (512 B), packed row-major into LDS.
__device__ __forceinline__ void tdm_load_tile_128x32(const _Float16* gsrc,
                                                     unsigned ldsoff) {
    unsigned long long ga = (unsigned long long)(size_t)gsrc;
    // D# group 0: [1:0]=count=1 | lds_addr | global_addr[56:0] | type=2
    v4u g0 = { 1u,
               ldsoff,
               (unsigned)ga,
               ((unsigned)((ga >> 32) & 0x01FFFFFFull)) | 0x80000000u };
    // D# group 1 (see CDNA5 ISA):
    //  d0: wg_mask=0, data_size=1 (2 B), no barrier/iterate/pad
    //  d1: [31:16] = tensor_dim0[15:0] = 256
    //  d2: [15:0]  = tensor_dim0[31:16]=0, [31:16]=tensor_dim1[15:0]=0
    //  d3: [15:0]  = tensor_dim1[31:16]=0x10 (dim1=1M, no OOB), [31:16]=tile_dim0=32
    //  d4: [15:0]  = tile_dim1=128, [31:16]=tile_dim2=0
    //  d5: tensor_dim0_stride[31:0] = 256
    //  d6,d7: stride hi / dim1 stride = 0 (2-D tile)
    v8i g1 = { 0x00010000,
               (int)(256u << 16),
               0,
               (int)(0x10u | (32u << 16)),
               128,
               256,
               0, 0 };
    v4i z4 = { 0, 0, 0, 0 };             // groups 2/3 unused for 2-D tiles
    v8i z8 = { 0, 0, 0, 0, 0, 0, 0, 0 }; // extra arg in 6-arg builtin form
    __builtin_amdgcn_tensor_load_to_lds(g0, g1, z4, z4, z8, 0);
}
#endif

__global__ __launch_bounds__(256) void gemm_l_kernel(const _Float16* __restrict__ Qh,
                                                     const _Float16* __restrict__ Dh,
                                                     float* __restrict__ L) {
    const int b  = blockIdx.z;
    const int n0 = blockIdx.x * TN;
    const int m0 = blockIdx.y * TM;

    __shared__ _Float16 As[2][TM][TK];   // 8 KB per buffer
    __shared__ _Float16 Bs[2][TN][TK];   // Dn rows row-major == B columns

    const _Float16* Ab = Qh + (size_t)b * NQ * DIM + (size_t)m0 * DIM;
    const _Float16* Bb = Dh + (size_t)b * NC * DIM + (size_t)n0 * DIM;

    const int t    = threadIdx.x;
    const int lane = t & 31;
    const int wave = t >> 5;
    const int wm   = wave & 1;   // 64-row block
    const int wn   = wave >> 1;  // 32-col block

    v8f acc[4][2] = {};

    // ---- stage k-slab 0 into buffer 0 ----
#if USE_TDM
    if (wave == 0)
        tdm_load_tile_128x32(Ab, (unsigned)(size_t)&As[0][0][0]);
    else if (wave == 1)
        tdm_load_tile_128x32(Bb, (unsigned)(size_t)&Bs[0][0][0]);
    if (wave < 2)
        __builtin_amdgcn_s_wait_tensorcnt(0);
#else
    #pragma unroll
    for (int i = 0; i < 2; ++i) {
        int idx = t + i * 256;
        int r = idx >> 2, c = idx & 3;
        *(v4i*)(&As[0][r][c * 8]) = *(const v4i*)(Ab + (size_t)r * DIM + c * 8);
        *(v4i*)(&Bs[0][r][c * 8]) = *(const v4i*)(Bb + (size_t)r * DIM + c * 8);
    }
#endif
    __syncthreads();

    const int koff = (lane & 16) ? 8 : 0;   // lanes 16..31: K 8..15 / 24..31
    const int lrow = lane & 15;

    union FH { v4i i2[2]; v16h h; };

    #pragma unroll
    for (int kk = 0; kk < 8; ++kk) {
        const int buf = kk & 1;
        if (kk < 7) {
            const int k0 = (kk + 1) * TK;
            const int nb = buf ^ 1;
#if USE_TDM
            if (wave == 0)
                tdm_load_tile_128x32(Ab + k0, (unsigned)(size_t)&As[nb][0][0]);
            else if (wave == 1)
                tdm_load_tile_128x32(Bb + k0, (unsigned)(size_t)&Bs[nb][0][0]);
#else
            #pragma unroll
            for (int i = 0; i < 2; ++i) {
                int idx = t + i * 256;
                int r = idx >> 2, c = idx & 3;
                *(v4i*)(&As[nb][r][c * 8]) =
                    *(const v4i*)(Ab + (size_t)r * DIM + k0 + c * 8);
                *(v4i*)(&Bs[nb][r][c * 8]) =
                    *(const v4i*)(Bb + (size_t)r * DIM + k0 + c * 8);
            }
#endif
        }

        FH afrag[4], bfrag[2];
        #pragma unroll
        for (int i = 0; i < 4; ++i) {
            const _Float16* p = &As[buf][wm * 64 + i * 16 + lrow][koff];
            afrag[i].i2[0] = *(const v4i*)(p);
            afrag[i].i2[1] = *(const v4i*)(p + 16);
        }
        #pragma unroll
        for (int j = 0; j < 2; ++j) {
            const _Float16* p = &Bs[buf][wn * 32 + j * 16 + lrow][koff];
            bfrag[j].i2[0] = *(const v4i*)(p);
            bfrag[j].i2[1] = *(const v4i*)(p + 16);
        }

        #pragma unroll
        for (int i = 0; i < 4; ++i)
            #pragma unroll
            for (int j = 0; j < 2; ++j)
                acc[i][j] = __builtin_amdgcn_wmma_f32_16x16x32_f16(
                    false, afrag[i].h, false, bfrag[j].h,
                    (short)0, acc[i][j], false, false);

#if USE_TDM
        if (wave < 2)
            __builtin_amdgcn_s_wait_tensorcnt(0);
#endif
        __syncthreads();
    }

    // C/D layout: VGPR r -> M = r (lanes 0-15) or r+8 (lanes 16-31); N = lane%16
    float* Lb = L + (size_t)b * NQ * NC;
    const int nloc = lane & 15;
    const int mhi  = (lane >> 4) & 1;
    #pragma unroll
    for (int i = 0; i < 4; ++i) {
        const int mBase = m0 + wm * 64 + i * 16 + mhi * 8;
        #pragma unroll
        for (int j = 0; j < 2; ++j) {
            const int c = n0 + wn * 32 + j * 16 + nloc;
            #pragma unroll
            for (int r = 0; r < 8; ++r)
                Lb[(size_t)(mBase + r) * NC + c] = acc[i][j][r];
        }
    }
}

// ---------------------------------------------------------------------------
// 3) Column softmax: A_D[b,c,q] = softmax_q(L[b,q,c]), written transposed.
// ---------------------------------------------------------------------------
__global__ __launch_bounds__(256) void colsoftmax_kernel(const float* __restrict__ L,
                                                         float* __restrict__ AD) {
    const int b  = blockIdx.y;
    const int c0 = blockIdx.x * 32;
    const int t  = threadIdx.x;
    const int tx = t & 31;
    const int ty = t >> 5;

    const float* Lb = L + (size_t)b * NQ * NC;

    float vals[64];
    float m = -1e30f;
    #pragma unroll
    for (int i = 0; i < 64; ++i) {
        vals[i] = Lb[(size_t)(ty + 8 * i) * NC + c0 + tx];
        m = fmaxf(m, vals[i]);
    }

    __shared__ float red[8][32];
    red[ty][tx] = m;
    __syncthreads();
    if (ty == 0) {
        float mm = red[0][tx];
        #pragma unroll
        for (int k = 1; k < 8; ++k) mm = fmaxf(mm, red[k][tx]);
        red[0][tx] = mm;
    }
    __syncthreads();
    const float cmax = red[0][tx];
    __syncthreads();

    float s = 0.f;
    #pragma unroll
    for (int i = 0; i < 64; ++i) {
        vals[i] = __expf(vals[i] - cmax);
        s += vals[i];
    }
    red[ty][tx] = s;
    __syncthreads();
    if (ty == 0) {
        float ss = 0.f;
        #pragma unroll
        for (int k = 0; k < 8; ++k) ss += red[k][tx];
        red[0][tx] = 1.0f / ss;
    }
    __syncthreads();
    const float inv = red[0][tx];

    __shared__ float T[32][9];
    float* ADb = AD + (size_t)b * NC * NQ;
    const int cl = t >> 3, ql = t & 7;
    #pragma unroll
    for (int i = 0; i < 64; ++i) {
        __syncthreads();
        T[tx][ty] = vals[i] * inv;                    // (c0+tx, q = 8i+ty)
        __syncthreads();
        ADb[(size_t)(c0 + cl) * NQ + 8 * i + ql] = T[cl][ql];
    }
}

// ---------------------------------------------------------------------------
// 4) Row softmax in place: A_Q[b,q,:] = softmax(L[b,q,:]) over 4096 entries.
// ---------------------------------------------------------------------------
__global__ __launch_bounds__(256) void rowsoftmax_kernel(float* __restrict__ AQ) {
    float* p = AQ + (size_t)blockIdx.x * NC;
    const int t = threadIdx.x;
    const int lane = t & 31, wv = t >> 5;

    float vals[16];
    float m = -1e30f;
    #pragma unroll
    for (int i = 0; i < 16; ++i) {
        vals[i] = p[t + 256 * i];
        m = fmaxf(m, vals[i]);
    }
    #pragma unroll
    for (int off = 16; off >= 1; off >>= 1)
        m = fmaxf(m, __shfl_xor(m, off, 32));

    __shared__ float red[8];
    if (lane == 0) red[wv] = m;
    __syncthreads();
    float bm = red[0];
    #pragma unroll
    for (int k = 1; k < 8; ++k) bm = fmaxf(bm, red[k]);

    float s = 0.f;
    #pragma unroll
    for (int i = 0; i < 16; ++i) {
        vals[i] = __expf(vals[i] - bm);
        s += vals[i];
    }
    __syncthreads();
    #pragma unroll
    for (int off = 16; off >= 1; off >>= 1)
        s += __shfl_xor(s, off, 32);
    if (lane == 0) red[wv] = s;
    __syncthreads();
    float bs = red[0];
    #pragma unroll
    for (int k = 1; k < 8; ++k) bs += red[k];
    const float inv = 1.0f / bs;

    #pragma unroll
    for (int i = 0; i < 16; ++i)
        p[t + 256 * i] = vals[i] * inv;
}

// ---------------------------------------------------------------------------
// 5) Means.
// ---------------------------------------------------------------------------
__global__ __launch_bounds__(256) void aqvec_kernel(const float* __restrict__ AQ,
                                                    float* __restrict__ AQv) {
    const int b = blockIdx.y;
    const int c = blockIdx.x * 256 + threadIdx.x;
    const float* p = AQ + (size_t)b * NQ * NC + c;
    float s = 0.f;
    for (int q = 0; q < NQ; ++q) s += p[(size_t)q * NC];
    AQv[b * NC + c] = s * (1.0f / (float)NQ);
}

__global__ __launch_bounds__(256) void zero_kernel(float* __restrict__ p, int n) {
    const int i = blockIdx.x * 256 + threadIdx.x;
    if (i < n) p[i] = 0.f;
}

__global__ __launch_bounds__(256) void advec_kernel(const float* __restrict__ AD,
                                                    float* __restrict__ ADv) {
    const int b = blockIdx.y;
    const int t = threadIdx.x;
    const float* base = AD + (size_t)b * NC * NQ + (size_t)blockIdx.x * 256 * NQ;
    float s0 = 0.f, s1 = 0.f;
    for (int c = 0; c < 256; ++c) {
        s0 += base[(size_t)c * NQ + t];
        s1 += base[(size_t)c * NQ + t + 256];
    }
    atomicAdd(&ADv[b * NQ + t],       s0 * (1.0f / (float)NC));
    atomicAdd(&ADv[b * NQ + t + 256], s1 * (1.0f / (float)NC));
}

// ---------------------------------------------------------------------------
extern "C" void kernel_launch(void* const* d_in, const int* in_sizes, int n_in,
                              void* d_out, int out_size, void* d_ws, size_t ws_size,
                              hipStream_t stream) {
    const float* Q  = (const float*)d_in[0];   // [16,512,256]
    const float* Dc = (const float*)d_in[1];   // [16,4096,256]

    float* out = (float*)d_out;
    float* AQ  = out;                                   // 16*512*4096
    float* AD  = out + (size_t)BATCH * NQ * NC;         // 16*4096*512
    float* AQv = AD  + (size_t)BATCH * NC * NQ;         // 16*4096
    float* ADv = AQv + (size_t)BATCH * NC;              // 16*512

    _Float16* Qh = (_Float16*)d_ws;                     // 4 MB
    _Float16* Dh = Qh + (size_t)BATCH * NQ * DIM;       // 32 MB

    normalize_kernel<<<(BATCH * NQ) / 8, 256, 0, stream>>>(Q, Qh, BATCH * NQ);
    normalize_kernel<<<(BATCH * NC) / 8, 256, 0, stream>>>(Dc, Dh, BATCH * NC);

    // L = Qn * Dn^T -> written into the A_Q region of d_out
    gemm_l_kernel<<<dim3(NC / TN, NQ / TM, BATCH), 256, 0, stream>>>(Qh, Dh, AQ);

    // Column softmax first (consumes L), then row softmax in place.
    colsoftmax_kernel<<<dim3(NC / 32, BATCH), 256, 0, stream>>>(AQ, AD);
    rowsoftmax_kernel<<<BATCH * NQ, 256, 0, stream>>>(AQ);

    aqvec_kernel<<<dim3(NC / 256, BATCH), 256, 0, stream>>>(AQ, AQv);
    zero_kernel<<<(BATCH * NQ + 255) / 256, 256, 0, stream>>>(ADv, BATCH * NQ);
    advec_kernel<<<dim3(NC / 256, BATCH), 256, 0, stream>>>(AD, ADv);
}
